// attention_82789789597757
// MI455X (gfx1250) — compile-verified, matrix operand order checked by reading, not run on hardware
//
#include <hip/hip_runtime.h>
#include <hip/hip_bf16.h>
#include <math.h>

// ---------------------------------------------------------------------------
// CDNA5 / gfx1250 pooled-attention transformer block.
//  - all GEMMs via v_wmma_f32_16x16x32_bf16 (wave32 WMMA, f32 accumulate)
//  - GEMM tiles streamed with global_load_async_to_lds_b128 (ASYNCcnt),
//    3-stage pipeline, one barrier per K-step
//  - softmax reductions via DPP16 butterflies (no ds_bpermute stalls)
// ---------------------------------------------------------------------------

typedef __attribute__((ext_vector_type(16))) __bf16 v16bf;
typedef __attribute__((ext_vector_type(8)))  float  v8f;

union FragU { v16bf v; uint4 q[2]; };

#define BB     16
#define HH     56
#define WWID   56
#define CC     256
#define HWTOK  (HH * WWID)        // 3136
#define MROWS  (BB * HWTOK)       // 50176
#define NHEAD  8
#define HDIM   16
#define KPOOL  7
#define QROWS  (BB * KPOOL * KPOOL)  // 784
#define XO_ELEMS ((long)MROWS * CC)  // 12845056 floats (first output)

__device__ __forceinline__ float gelu_exact(float x) {
  return 0.5f * x * (1.0f + erff(x * 0.70710678118654752440f));
}

__device__ __forceinline__ v8f wmma_bf16(v16bf a, v16bf b, v8f c) {
  return __builtin_amdgcn_wmma_f32_16x16x32_bf16(
      /*neg_a=*/false, a, /*neg_b=*/false, b,
      /*c_mod=*/(short)0, c, /*reuse_a=*/false, /*reuse_b=*/false);
}

__device__ __forceinline__ uint4 zero4() { uint4 z{0u, 0u, 0u, 0u}; return z; }

// ---- async memory->LDS copy (gfx1250), tracked by ASYNCcnt ----------------
__device__ __forceinline__ void async_copy_b128(const void* gptr, void* lptr) {
  unsigned lds = (unsigned)(uintptr_t)lptr;  // flat->LDS aperture: low 32 bits
  unsigned long long ga = (unsigned long long)(uintptr_t)gptr;
  asm volatile("global_load_async_to_lds_b128 %0, %1, off"
               :: "v"(lds), "v"(ga) : "memory");
}
__device__ __forceinline__ void wait_async_le2() {
  asm volatile("s_wait_asynccnt 0x2" ::: "memory");
}
__device__ __forceinline__ void wait_async_0() {
  asm volatile("s_wait_asynccnt 0x0" ::: "memory");
}

// ---- DPP16 butterfly reductions over 16-lane groups (no LDS traffic) ------
template <int CTRL>
__device__ __forceinline__ float dpp_f(float x) {
  int xi = __builtin_bit_cast(int, x);
  int r = __builtin_amdgcn_update_dpp(xi, xi, CTRL, 0xf, 0xf, true);
  return __builtin_bit_cast(float, r);
}
__device__ __forceinline__ float red_max16(float x) {
  x = fmaxf(x, dpp_f<0xB1>(x));   // quad_perm [1,0,3,2]  (xor 1)
  x = fmaxf(x, dpp_f<0x4E>(x));   // quad_perm [2,3,0,1]  (xor 2)
  x = fmaxf(x, dpp_f<0x141>(x));  // row_half_mirror      (within 8)
  x = fmaxf(x, dpp_f<0x140>(x));  // row_mirror           (within 16)
  return x;
}
__device__ __forceinline__ float red_sum16(float x) {
  x += dpp_f<0xB1>(x);
  x += dpp_f<0x4E>(x);
  x += dpp_f<0x141>(x);
  x += dpp_f<0x140>(x);
  return x;
}

// ---------------------------------------------------------------------------
// Weight convert + transpose: W[K][N] f32 -> Wt[N][K] bf16.
// ---------------------------------------------------------------------------
__global__ __launch_bounds__(256)
void k_wt(const float* __restrict__ w, __bf16* __restrict__ wt, int K, int N) {
  int idx = blockIdx.x * 256 + threadIdx.x;
  if (idx >= K * N) return;
  int n = idx % N, k = idx / N;
  wt[(size_t)n * K + k] = (__bf16)w[(size_t)k * N + n];
}

// ---------------------------------------------------------------------------
// LayerNorm over C=256, one wave per row, bf16 out.
// ---------------------------------------------------------------------------
__global__ __launch_bounds__(256)
void k_layernorm(const float* __restrict__ x, const float* __restrict__ w,
                 const float* __restrict__ b, __bf16* __restrict__ out, long M) {
  long row = (long)blockIdx.x * 8 + (threadIdx.x >> 5);
  int lane = threadIdx.x & 31;
  if (row >= M) return;
  const float* xr = x + row * CC;
  float4 a0 = *(const float4*)(xr + lane * 8);
  float4 a1 = *(const float4*)(xr + lane * 8 + 4);
  float v[8] = {a0.x, a0.y, a0.z, a0.w, a1.x, a1.y, a1.z, a1.w};
  float s = 0.f, s2 = 0.f;
#pragma unroll
  for (int k = 0; k < 8; ++k) { s += v[k]; s2 += v[k] * v[k]; }
  // 32-lane reduction: 16-lane DPP butterfly + cross-half via permlane-style shfl
  s  = red_sum16(s);
  s2 = red_sum16(s2);
  s  += __shfl_xor(s, 16, 32);
  s2 += __shfl_xor(s2, 16, 32);
  float mean = s * (1.0f / CC);
  float var  = s2 * (1.0f / CC) - mean * mean;
  float inv  = rsqrtf(var + 1e-6f);
  union { __bf16 h[8]; uint4 u; } ov;
#pragma unroll
  for (int k = 0; k < 8; ++k) {
    int c = lane * 8 + k;
    ov.h[k] = (__bf16)((v[k] - mean) * inv * w[c] + b[c]);
  }
  *(uint4*)(out + row * CC + lane * 8) = ov.u;
}

// ---------------------------------------------------------------------------
// Adaptive 8x8 mean pool: xn[B,56,56,C] bf16 -> pooled[B*49, C] bf16.
// ---------------------------------------------------------------------------
__global__ __launch_bounds__(256)
void k_pool(const __bf16* __restrict__ xn, __bf16* __restrict__ pooled) {
  int idx = blockIdx.x * 256 + threadIdx.x;
  if (idx >= QROWS * CC) return;
  int c = idx & 255;
  int p = idx >> 8;
  int kx = p % KPOOL; p /= KPOOL;
  int ky = p % KPOOL;
  int b  = p / KPOOL;
  float s = 0.f;
#pragma unroll 4
  for (int iy = 0; iy < 8; ++iy)
    for (int ix = 0; ix < 8; ++ix)
      s += (float)xn[(((long)b * HH + ky * 8 + iy) * WWID + kx * 8 + ix) * CC + c];
  pooled[idx] = (__bf16)(s * (1.0f / 64.0f));
}

// ---------------------------------------------------------------------------
// bf16 WMMA GEMM: C[M,N] = epi(A[M,K] @ Wt[N,K]^T + bias).
// 128x128 block tile, BK=32, 8 waves (4M x 2N), 2x4 WMMA accs per wave.
// Tiles streamed via global_load_async_to_lds_b128, 3-stage pipeline,
// one s_barrier per K-step. OOB M rows clamped (masked at store).
// EPI: 0 = gelu->bf16, 1 = bf16, 2 = f32, 3 = f32 (scale[n]*x + res).
// ---------------------------------------------------------------------------
template <int EPI>
__global__ __launch_bounds__(256)
void k_gemm(const __bf16* __restrict__ A, const __bf16* __restrict__ Bt,
            int M, int N, int K,
            const float* __restrict__ bias, const float* __restrict__ scale,
            const float* __restrict__ res, void* __restrict__ out) {
  constexpr int LDT = 40;                       // padded LDS stride (elems)
  constexpr int NBUF = 3;                       // 3-stage async pipeline
  __shared__ __bf16 As[NBUF][128 * LDT];
  __shared__ __bf16 Bs[NBUF][128 * LDT];

  int tid  = threadIdx.x;
  int lane = tid & 31, wave = tid >> 5;
  int wm = wave & 3, wn = wave >> 2;            // 4 waves in M, 2 in N
  int half = lane >> 4, l15 = lane & 15;
  long blockM = (long)blockIdx.x * 128;
  long blockN = (long)blockIdx.y * 128;

  v8f acc[2][4];
  const v8f vz = {0.f, 0.f, 0.f, 0.f, 0.f, 0.f, 0.f, 0.f};
#pragma unroll
  for (int i = 0; i < 2; ++i)
#pragma unroll
    for (int j = 0; j < 4; ++j) acc[i][j] = vz;

  // staging role of this thread: one b128 of A + one b128 of Bt per stage
  int lrow = tid >> 1;                          // 0..127
  int lcol = (tid & 1) * 16;                    // 0 or 16 (elems)
  long arow = blockM + lrow;
  if (arow >= M) arow = M - 1;                  // clamp; garbage rows masked
  const __bf16* agp = A  + arow * (size_t)K + lcol;
  const __bf16* bgp = Bt + (blockN + lrow) * (size_t)K + lcol;
  int lidx = lrow * LDT + lcol;

  int nk = K >> 5;
  async_copy_b128(agp, &As[0][lidx]);           // prologue: stage 0 in flight
  async_copy_b128(bgp, &Bs[0][lidx]);

  for (int it = 0; it < nk; ++it) {
    if (it + 1 < nk) {                          // prefetch stage it+1
      int nb = (it + 1) % NBUF;
      async_copy_b128(agp + (size_t)(it + 1) * 32, &As[nb][lidx]);
      async_copy_b128(bgp + (size_t)(it + 1) * 32, &Bs[nb][lidx]);
      wait_async_le2();                         // drain stage it
    } else {
      wait_async_0();
    }
    __syncthreads();                            // tile it visible to all waves

    int cb = it % NBUF;
    FragU af[2], bf[4];
#pragma unroll
    for (int i = 0; i < 2; ++i) {
      int r = (wm * 2 + i) * 16 + l15;
      af[i].q[0] = *(const uint4*)&As[cb][r * LDT + half * 8];
      af[i].q[1] = *(const uint4*)&As[cb][r * LDT + half * 8 + 16];
    }
#pragma unroll
    for (int j = 0; j < 4; ++j) {
      int n = (wn * 4 + j) * 16 + l15;
      bf[j].q[0] = *(const uint4*)&Bs[cb][n * LDT + half * 8];
      bf[j].q[1] = *(const uint4*)&Bs[cb][n * LDT + half * 8 + 16];
    }
#pragma unroll
    for (int i = 0; i < 2; ++i)
#pragma unroll
      for (int j = 0; j < 4; ++j)
        acc[i][j] = wmma_bf16(af[i].v, bf[j].v, acc[i][j]);
  }

  // epilogue (C layout: VGPR v -> row v (lanes 0-15) / v+8 (lanes 16-31))
#pragma unroll
  for (int i = 0; i < 2; ++i) {
#pragma unroll
    for (int j = 0; j < 4; ++j) {
#pragma unroll
      for (int v = 0; v < 8; ++v) {
        long rg = blockM + (wm * 2 + i) * 16 + v + half * 8;
        long cg = blockN + (wn * 4 + j) * 16 + l15;
        if (rg < M) {
          float x = acc[i][j][v] + bias[cg];
          size_t o = (size_t)rg * N + cg;
          if constexpr (EPI == 0) {
            ((__bf16*)out)[o] = (__bf16)gelu_exact(x);
          } else if constexpr (EPI == 1) {
            ((__bf16*)out)[o] = (__bf16)x;
          } else if constexpr (EPI == 2) {
            ((float*)out)[o] = x;
          } else {  // layer-scale + residual
            ((float*)out)[o] = scale[cg] * x + res[o];
          }
        }
      }
    }
  }
}

// ---------------------------------------------------------------------------
// Flash attention, one workgroup per (b, head): 49 queries x 3136 keys x 16.
// QK^T and P*V via WMMA bf16 (head dim 16 zero-padded to K=32).
// Softmax stats via DPP16 butterflies; P relayouted through LDS.
// ---------------------------------------------------------------------------
__global__ __launch_bounds__(128)
void k_attention(const float* __restrict__ qbuf,    // [784, 128] f32
                 const __bf16* __restrict__ kvb,    // [50176, 256] bf16 (k|v)
                 float* __restrict__ headout) {     // [B, NH, 49, 16] f32
  constexpr int LDT = 40;
  __shared__ __bf16 mq[64 * 16];
  __shared__ __bf16 vT[16 * LDT];                   // v chunk [hd][key]
  __shared__ __bf16 Pl[4 * 16 * LDT];               // per-wave P staging

  int bh = blockIdx.x;
  int b = bh >> 3, nh = bh & 7;
  int tid = threadIdx.x;
  int lane = tid & 31, wave = tid >> 5;
  int half = lane >> 4, l15 = lane & 15;

  for (int e = tid; e < 64 * 16; e += 128) {        // stage scaled Q
    int row = e >> 4, hd = e & 15;
    float v = 0.0f;
    if (row < 49) v = qbuf[(b * 49 + row) * 128 + nh * 16 + hd] * 0.25f;
    mq[e] = (__bf16)v;
  }
  __syncthreads();

  FragU mf;
  mf.q[0] = *(const uint4*)&mq[(wave * 16 + l15) * 16 + half * 8];
  mf.q[1] = zero4();                                // K=16..31 padding

  v8f acc = {0.f, 0.f, 0.f, 0.f, 0.f, 0.f, 0.f, 0.f};
  float mrow[8], lrow[8];
#pragma unroll
  for (int v = 0; v < 8; ++v) { mrow[v] = -1e30f; lrow[v] = 0.0f; }

  __bf16* Pw = &Pl[wave * 16 * LDT];

  for (int ch = 0; ch < HWTOK / 32; ++ch) {         // 98 chunks of 32 keys
    int hw0 = ch * 32;
    __syncthreads();                                // prev vT readers done
    for (int e = tid; e < 512; e += 128) {          // stage V chunk transposed
      int key = e & 31, hd = e >> 5;
      vT[hd * LDT + key] =
          kvb[((long)(b * HWTOK + hw0 + key)) * CC + 128 + nh * 16 + hd];
    }

    FragU kf0, kf1;                                 // K fragments from global
    {
      const __bf16* kp0 =
          kvb + ((long)(b * HWTOK + hw0 + l15)) * CC + nh * 16 + half * 8;
      const __bf16* kp1 =
          kvb + ((long)(b * HWTOK + hw0 + 16 + l15)) * CC + nh * 16 + half * 8;
      kf0.q[0] = *(const uint4*)kp0; kf0.q[1] = zero4();
      kf1.q[0] = *(const uint4*)kp1; kf1.q[1] = zero4();
    }

    const v8f cz = {0.f, 0.f, 0.f, 0.f, 0.f, 0.f, 0.f, 0.f};
    v8f s0 = wmma_bf16(mf.v, kf0.v, cz);            // scores, keys 0..15
    v8f s1 = wmma_bf16(mf.v, kf1.v, cz);            // scores, keys 16..31

#pragma unroll
    for (int v = 0; v < 8; ++v) {                   // online softmax per row
      float a = s0[v], c2 = s1[v];
      float mx = red_max16(fmaxf(a, c2));           // DPP butterfly, no LDS
      float mnew  = fmaxf(mrow[v], mx);
      float alpha = __expf(mrow[v] - mnew);
      float p0 = __expf(a - mnew), p1 = __expf(c2 - mnew);
      float ps = red_sum16(p0 + p1);
      lrow[v] = lrow[v] * alpha + ps;
      mrow[v] = mnew;
      acc[v] *= alpha;
      int prow = v + half * 8;                      // C layout -> A layout
      Pw[prow * LDT + l15]      = (__bf16)p0;
      Pw[prow * LDT + 16 + l15] = (__bf16)p1;
    }
    __syncthreads();                                // vT + P visible

    FragU pf, vf;
    pf.q[0] = *(const uint4*)&Pw[l15 * LDT + half * 8];
    pf.q[1] = *(const uint4*)&Pw[l15 * LDT + half * 8 + 16];
    vf.q[0] = *(const uint4*)&vT[l15 * LDT + half * 8];
    vf.q[1] = *(const uint4*)&vT[l15 * LDT + half * 8 + 16];
    acc = wmma_bf16(pf.v, vf.v, acc);               // O += P * V
  }

#pragma unroll
  for (int v = 0; v < 8; ++v) {
    int qrow = wave * 16 + v + half * 8;
    if (qrow < 49)
      headout[((long)(b * NHEAD + nh) * 49 + qrow) * HDIM + l15] =
          acc[v] / lrow[v];
  }
}

// ---------------------------------------------------------------------------
// Bilinear 7->56 upsample (align_corners=False): headout -> up[B*HW,128] bf16.
// ---------------------------------------------------------------------------
__global__ __launch_bounds__(256)
void k_upsample(const float* __restrict__ ho, __bf16* __restrict__ up) {
  long idx = (long)blockIdx.x * 256 + threadIdx.x;
  if (idx >= (long)MROWS * 128) return;
  int c = idx & 127;
  long t = idx >> 7;
  int w = (int)(t % WWID); t /= WWID;
  int h = (int)(t % HH);
  int b = (int)(t / HH);
  int nh = c >> 4, hd = c & 15;
  float sy = (h + 0.5f) * 0.125f - 0.5f;
  float sx = (w + 0.5f) * 0.125f - 0.5f;
  int y0 = (int)floorf(sy), x0 = (int)floorf(sx);
  float wy = sy - y0, wx = sx - x0;
  int y0c = max(y0, 0), y1c = min(y0 + 1, 6);
  int x0c = max(x0, 0), x1c = min(x0 + 1, 6);
  long base = ((long)(b * NHEAD + nh) * 49) * HDIM + hd;
  float v00 = ho[base + (y0c * 7 + x0c) * HDIM];
  float v01 = ho[base + (y0c * 7 + x1c) * HDIM];
  float v10 = ho[base + (y1c * 7 + x0c) * HDIM];
  float v11 = ho[base + (y1c * 7 + x1c) * HDIM];
  float val = (1.f - wy) * ((1.f - wx) * v00 + wx * v01) +
              wy * ((1.f - wx) * v10 + wx * v11);
  up[idx] = (__bf16)val;
}

// ---------------------------------------------------------------------------
// Depthwise 3x3 conv (zero pad) + residual + bias + exact GELU on 4C=1024.
// ---------------------------------------------------------------------------
__global__ __launch_bounds__(256)
void k_dwconv_gelu(const __bf16* __restrict__ y1, const float* __restrict__ pw,
                   const float* __restrict__ pb, __bf16* __restrict__ y2) {
  long idx = (long)blockIdx.x * 256 + threadIdx.x;
  if (idx >= (long)MROWS * 1024) return;
  int c = idx & 1023;
  long t = idx >> 10;
  int w = (int)(t % WWID); t /= WWID;
  int h = (int)(t % HH);
  int b = (int)(t / HH);
  float conv = 0.f;
#pragma unroll
  for (int ky = 0; ky < 3; ++ky) {
    int hh = h + ky - 1;
    if ((unsigned)hh >= HH) continue;
#pragma unroll
    for (int kx = 0; kx < 3; ++kx) {
      int ww = w + kx - 1;
      if ((unsigned)ww >= WWID) continue;
      conv += (float)y1[(((long)b * HH + hh) * WWID + ww) * 1024 + c] *
              pw[c * 9 + ky * 3 + kx];
    }
  }
  float val = conv + pb[c] + (float)y1[idx];
  y2[idx] = (__bf16)gelu_exact(val);
}

// ---------------------------------------------------------------------------
extern "C" void kernel_launch(void* const* d_in, const int* in_sizes, int n_in,
                              void* d_out, int out_size, void* d_ws, size_t ws_size,
                              hipStream_t stream) {
  (void)in_sizes; (void)n_in; (void)out_size; (void)ws_size;
  const float* x       = (const float*)d_in[0];
  // d_in[1] (x_e), d_in[4..5] (norme_*): dead code in the reference, skipped.
  const float* norm_w  = (const float*)d_in[2];
  const float* norm_b  = (const float*)d_in[3];
  const float* l_w     = (const float*)d_in[6];
  const float* l_b     = (const float*)d_in[7];
  const float* kv_w    = (const float*)d_in[8];
  const float* kv_b    = (const float*)d_in[9];
  const float* sc_w    = (const float*)d_in[10];
  const float* sc_b    = (const float*)d_in[11];
  const float* proj_w  = (const float*)d_in[12];
  const float* proj_b  = (const float*)d_in[13];
  const float* ls1     = (const float*)d_in[14];
  const float* ls2     = (const float*)d_in[15];
  const float* mnorm_w = (const float*)d_in[16];
  const float* mnorm_b = (const float*)d_in[17];
  const float* fc1_w   = (const float*)d_in[18];
  const float* fc1_b   = (const float*)d_in[19];
  const float* pos_w   = (const float*)d_in[20];
  const float* pos_b   = (const float*)d_in[21];
  const float* fc2_w   = (const float*)d_in[22];
  const float* fc2_b   = (const float*)d_in[23];

  float* out_xo = (float*)d_out;                 // [B,H,W,C]
  float* out_q  = (float*)d_out + XO_ELEMS;      // [B,7,7,128]

  char* p = (char*)d_ws;
  auto alloc = [&](size_t bytes) -> char* {
    char* r = p;
    p += (bytes + 255) & ~(size_t)255;
    return r;
  };
  __bf16* Wl    = (__bf16*)alloc((size_t)256 * 256 * 2);
  __bf16* Wkv   = (__bf16*)alloc((size_t)256 * 256 * 2);
  __bf16* Wsc   = (__bf16*)alloc((size_t)256 * 128 * 2);
  __bf16* Wproj = (__bf16*)alloc((size_t)128 * 256 * 2);
  __bf16* Wfc1  = (__bf16*)alloc((size_t)256 * 1024 * 2);
  __bf16* Wfc2  = (__bf16*)alloc((size_t)1024 * 256 * 2);
  __bf16* xn    = (__bf16*)alloc((size_t)MROWS * 256 * 2);
  __bf16* bact  = (__bf16*)alloc((size_t)MROWS * 256 * 2);
  __bf16* kvb   = (__bf16*)alloc((size_t)MROWS * 256 * 2);
  __bf16* pooled= (__bf16*)alloc((size_t)QROWS * 256 * 2);
  float*  hout  = (float*)alloc((size_t)BB * NHEAD * 49 * 16 * 4);
  __bf16* up    = (__bf16*)alloc((size_t)MROWS * 128 * 2);
  float*  xo    = (float*)alloc((size_t)MROWS * 256 * 4);
  __bf16* xoln  = (__bf16*)alloc((size_t)MROWS * 256 * 2);
  __bf16* y1    = (__bf16*)alloc((size_t)MROWS * 1024 * 2);
  __bf16* y2    = (__bf16*)alloc((size_t)MROWS * 1024 * 2);

  // 1) weights -> bf16, transposed to [N][K]
  k_wt<<<(256 * 256 + 255) / 256, 256, 0, stream>>>(l_w, Wl, 256, 256);
  k_wt<<<(256 * 256 + 255) / 256, 256, 0, stream>>>(kv_w, Wkv, 256, 256);
  k_wt<<<(256 * 128 + 255) / 256, 256, 0, stream>>>(sc_w, Wsc, 256, 128);
  k_wt<<<(128 * 256 + 255) / 256, 256, 0, stream>>>(proj_w, Wproj, 128, 256);
  k_wt<<<(256 * 1024 + 255) / 256, 256, 0, stream>>>(fc1_w, Wfc1, 256, 1024);
  k_wt<<<(1024 * 256 + 255) / 256, 256, 0, stream>>>(fc2_w, Wfc2, 1024, 256);

  // 2) xn = LN(x)
  k_layernorm<<<MROWS / 8, 256, 0, stream>>>(x, norm_w, norm_b, xn, MROWS);
  // 3) 8x8 mean pool
  k_pool<<<(QROWS * 256 + 255) / 256, 256, 0, stream>>>(xn, pooled);
  // 4) q = pooled @ sc_w + sc_b  (f32, straight into d_out's q slot)
  k_gemm<2><<<dim3((QROWS + 127) / 128, 128 / 128), 256, 0, stream>>>(
      pooled, Wsc, QROWS, 128, 256, sc_b, nullptr, nullptr, out_q);
  // 5) b_act = gelu(xn @ l_w + l_b)
  k_gemm<0><<<dim3(MROWS / 128, 256 / 128), 256, 0, stream>>>(
      xn, Wl, MROWS, 256, 256, l_b, nullptr, nullptr, bact);
  // 6) kv = b_act @ kv_w + kv_b
  k_gemm<1><<<dim3(MROWS / 128, 256 / 128), 256, 0, stream>>>(
      bact, Wkv, MROWS, 256, 256, kv_b, nullptr, nullptr, kvb);
  // 7) flash attention per (b, head)
  k_attention<<<BB * NHEAD, 128, 0, stream>>>(out_q, kvb, hout);
  // 8) bilinear upsample 7x7 -> 56x56
  k_upsample<<<(int)(((long)MROWS * 128 + 255) / 256), 256, 0, stream>>>(hout, up);
  // 9) xo = ls1 * (up @ proj_w + proj_b) + x
  k_gemm<3><<<dim3(MROWS / 128, 256 / 128), 256, 0, stream>>>(
      up, Wproj, MROWS, 256, 128, proj_b, ls1, x, xo);
  // 10) LN(xo)
  k_layernorm<<<MROWS / 8, 256, 0, stream>>>(xo, mnorm_w, mnorm_b, xoln, MROWS);
  // 11) y1 = xoln @ fc1_w + fc1_b
  k_gemm<1><<<dim3(MROWS / 128, 1024 / 128), 256, 0, stream>>>(
      xoln, Wfc1, MROWS, 1024, 256, fc1_b, nullptr, nullptr, y1);
  // 12) y2 = gelu(dwconv3x3(y1) + pos_b + y1)
  k_dwconv_gelu<<<(int)(((long)MROWS * 1024 + 255) / 256), 256, 0, stream>>>(
      y1, pos_w, pos_b, y2);
  // 13) out = xo + ls2 * (y2 @ fc2_w + fc2_b)
  k_gemm<3><<<dim3(MROWS / 128, 256 / 128), 256, 0, stream>>>(
      y2, Wfc2, MROWS, 256, 1024, fc2_b, ls2, xo, out_xo);
}
